// SimpleGatedLinearAttention_50740743635385
// MI455X (gfx1250) — matrix operand order repaired and checked by reading, not compiled, not run on hardware
//
#include <hip/hip_runtime.h>
#include <hip/hip_bf16.h>

typedef __bf16 bf16;
typedef __attribute__((ext_vector_type(16))) __bf16 v16bf;
typedef __attribute__((ext_vector_type(8)))  __bf16 v8bf;
typedef __attribute__((ext_vector_type(8)))  float  v8f;

#define WMMA_BF16(A, B, C) \
  __builtin_amdgcn_wmma_f32_16x16x32_bf16(false, (A), false, (B), (short)0, (C), false, false)

static __device__ __forceinline__ v16bf cat8(v8bf lo, v8bf hi) {
  v16bf r;
#pragma unroll
  for (int i = 0; i < 8; ++i) { r[i] = lo[i]; r[i + 8] = hi[i]; }
  return r;
}

// CDNA5 async global->LDS copy (16B), tracked by ASYNCcnt (bypasses VGPRs).
static __device__ __forceinline__ void async_g2l_b128(bf16* lds_ptr, const bf16* gptr) {
  unsigned int loff = (unsigned int)(size_t)lds_ptr;   // low 32 bits = LDS byte offset
  asm volatile("global_load_async_to_lds_b128 %0, %1, off"
               :: "v"(loff), "v"(gptr) : "memory");
}
static __device__ __forceinline__ void wait_async0() {
  asm volatile("s_wait_asynccnt 0x0" ::: "memory");
}

// ---------------------------------------------------------------------------
// fp32 -> bf16 elementwise convert
// ---------------------------------------------------------------------------
__global__ void f32_to_bf16_kernel(const float* __restrict__ in, bf16* __restrict__ out, int n) {
  int i = blockIdx.x * 256 + threadIdx.x;
  if (i < n) out[i] = (bf16)in[i];
}

// ---------------------------------------------------------------------------
// Tiled bf16 WMMA GEMM: C[M,N] = A[M,K] * B[K,N]; A,B bf16 row-major.
// Block = 256 threads (8 waves), tile 128x128, BK=32.
// Wave grid 2(m) x 4(n): each wave 64x32 region = 4x2 16x16 tiles.
// A tile staged with GLOBAL_LOAD_ASYNC_TO_LDS_B128 (no VGPR round-trip);
// B tile staged through VGPRs because it is transposed into LDS.
// STORE_F32 is a compile-time flag -> branch-free epilogue.
// ---------------------------------------------------------------------------
template <int STORE_F32>
__global__ __launch_bounds__(256)
void gemm_bf16_kernel(const bf16* __restrict__ A, const bf16* __restrict__ B,
                      void* __restrict__ Cout, int M, int N, int K) {
  const int LDA = 40;                    // padded strides (bf16 elems), rows 16B-aligned
  const int LDB = 40;
  __shared__ bf16 sA[128 * 40];          // A tile, row-major [m][k]
  __shared__ bf16 sB[128 * 40];          // B tile transposed [n][k]

  const int bm = blockIdx.y * 128;
  const int bn = blockIdx.x * 128;
  const int tid  = threadIdx.x;
  const int wave = tid >> 5;
  const int lane = tid & 31;
  const int half = lane >> 4;
  const int l16  = lane & 15;
  const int wm = (wave >> 2) * 64;       // 0 or 64
  const int wn = (wave & 3) * 32;        // 0,32,64,96

  v8f acc[4][2] = {};

  for (int k0 = 0; k0 < K; k0 += 32) {
    __syncthreads();
    // stage A asynchronously: 128x32 (4096 elems), 8 contiguous bf16 per copy
#pragma unroll
    for (int r = 0; r < 2; ++r) {
      int idx = (tid + r * 256) * 8;
      int row = idx >> 5, col = idx & 31;
      const bf16* gp = A + (size_t)(bm + row) * K + k0 + col;
      async_g2l_b128(sA + row * LDA + col, gp);
      __builtin_prefetch(gp + 32, 0, 1);           // speculative: next K tile
    }
    // stage B transposed: read 8 contiguous n, scatter into sB[n][k]
#pragma unroll
    for (int r = 0; r < 2; ++r) {
      int idx = (tid + r * 256) * 8;     // over 32x128
      int krow = idx >> 7, ncol = idx & 127;
      const bf16* gp = B + (size_t)(k0 + krow) * N + bn + ncol;
      v8bf v = *(const v8bf*)gp;
      __builtin_prefetch(gp + (size_t)32 * N, 0, 1);
#pragma unroll
      for (int e = 0; e < 8; ++e) sB[(ncol + e) * LDB + krow] = v[e];
    }
    wait_async0();
    __syncthreads();

    v16bf af[4], bfm[2];
#pragma unroll
    for (int mi = 0; mi < 4; ++mi) {
      const bf16* p = sA + (wm + mi * 16 + l16) * LDA + half * 8;
      af[mi] = cat8(*(const v8bf*)p, *(const v8bf*)(p + 16));
    }
#pragma unroll
    for (int ni = 0; ni < 2; ++ni) {
      const bf16* p = sB + (wn + ni * 16 + l16) * LDB + half * 8;
      bfm[ni] = cat8(*(const v8bf*)p, *(const v8bf*)(p + 16));
    }
#pragma unroll
    for (int mi = 0; mi < 4; ++mi)
#pragma unroll
      for (int ni = 0; ni < 2; ++ni)
        acc[mi][ni] = WMMA_BF16(af[mi], bfm[ni], acc[mi][ni]);
  }

  // branch-free epilogue: row base hoisted, per-element store is base + r*N
#pragma unroll
  for (int mi = 0; mi < 4; ++mi)
#pragma unroll
    for (int ni = 0; ni < 2; ++ni) {
      int col  = bn + wn + ni * 16 + l16;
      size_t base = (size_t)(bm + wm + mi * 16 + half * 8) * N + col;
#pragma unroll
      for (int r = 0; r < 8; ++r) {
        float v = acc[mi][ni][r];
        if (STORE_F32) ((float*)Cout)[base + (size_t)r * N] = v;
        else           ((bf16*)Cout)[base + (size_t)r * N] = (bf16)v;
      }
    }
}

// ---------------------------------------------------------------------------
// gk projection: gk[(b*4+h)*2048 + n] = logsigmoid(x . Wgk[:,h] + bgk[h]) / 16
// one block per token, 256 threads
// ---------------------------------------------------------------------------
__global__ __launch_bounds__(256)
void gk_proj_kernel(const float* __restrict__ x, const float* __restrict__ Wgk,
                    const float* __restrict__ bgk, float* __restrict__ gkb) {
  __shared__ float red[256];
  int t = blockIdx.x;                    // token 0..8191
  int tid = threadIdx.x;
  float p[4] = {0.f, 0.f, 0.f, 0.f};
  for (int i = tid; i < 1024; i += 256) {
    float xv = x[(size_t)t * 1024 + i];
#pragma unroll
    for (int h = 0; h < 4; ++h) p[h] += xv * Wgk[i * 4 + h];
  }
#pragma unroll
  for (int h = 0; h < 4; ++h) {
    red[tid] = p[h];
    __syncthreads();
    for (int s = 128; s > 0; s >>= 1) {
      if (tid < s) red[tid] += red[tid + s];
      __syncthreads();
    }
    if (tid == 0) {
      float u = red[0] + bgk[h];
      float ls = fminf(u, 0.f) - log1pf(__expf(-fabsf(u)));
      int b = t >> 11, n = t & 2047;
      gkb[((size_t)(b * 4 + h)) * 2048 + n] = ls * (1.0f / 16.0f);
    }
    __syncthreads();
  }
}

// ---------------------------------------------------------------------------
// Chunked simple-GLA core.  One block per (dv-tile=64, head, batch).
// 256 threads = 8 waves.  State S kept in LDS as bf16 column-major [dv][dk].
// ---------------------------------------------------------------------------
__global__ __launch_bounds__(256)
void gla_attn_kernel(const bf16* __restrict__ qb, const bf16* __restrict__ kb,
                     const bf16* __restrict__ vb, const float* __restrict__ gkb,
                     float* __restrict__ ob) {
  extern __shared__ char smem[];
  bf16*  sQ  = (bf16*)smem;                       // qhat [64][264] row-major (t,d)
  bf16*  sKT = sQ + 64 * 264;                     // khat^T [256][72]  (d,s)
  bf16*  sVT = sKT + 256 * 72;                    // v^T [64][72]      (dv,s)
  bf16*  sAb = sVT + 64 * 72;                     // masked scores [64][72] (t,s)
  bf16*  sS  = sAb + 64 * 72;                     // state^T [64 dv][264 dk]
  float* sGc = (float*)(sS + 64 * 264);           // gc[0..63], gt at [64]

  const int vt = blockIdx.x;                      // dv tile (0..7), 64 wide
  const int h  = blockIdx.y;
  const int b  = blockIdx.z;
  const int tid  = threadIdx.x;
  const int wave = tid >> 5;
  const int lane = tid & 31;
  const int half = lane >> 4;
  const int l16  = lane & 15;
  const size_t tok0 = (size_t)b * 2048;
  const float scale = 0.0625f;                    // 256^-0.5

  for (int i = tid; i < 64 * 264; i += 256) sS[i] = (bf16)0.f;

  for (int c = 0; c < 32; ++c) {
    __syncthreads();                              // prev chunk fully done
    if (tid == 0) {
      float a = 0.f;
      const float* gp = gkb + ((size_t)(b * 4 + h)) * 2048 + c * 64;
      for (int t = 0; t < 64; ++t) { a += gp[t]; sGc[t] = a; }
      sGc[64] = a;                                // gt
    }
    __syncthreads();

    // stage qhat = q*scale*exp(gc_t) and khat^T with khat = k*exp(-gc_s)
#pragma unroll
    for (int r = 0; r < 8; ++r) {
      int idx = tid + r * 256;                    // 0..2047 over 64x(256/8)
      int row = idx >> 5;                         // t
      int col = (idx & 31) * 8;                   // d
      size_t g = (tok0 + c * 64 + row) * (size_t)1024 + h * 256 + col;
      v8bf qv = *(const v8bf*)(qb + g);
      v8bf kv = *(const v8bf*)(kb + g);
      float eq = scale * __expf(sGc[row]);
      float ek = __expf(-sGc[row]);
      v8bf qh;
#pragma unroll
      for (int e = 0; e < 8; ++e) {
        qh[e] = (bf16)((float)qv[e] * eq);
        sKT[(col + e) * 72 + row] = (bf16)((float)kv[e] * ek);
      }
      *(v8bf*)(sQ + row * 264 + col) = qh;
    }
    // stage v^T (dv-tile of 64)
#pragma unroll
    for (int r = 0; r < 2; ++r) {
      int idx = tid + r * 256;                    // 0..511 over 64x(64/8)
      int row = idx >> 3;                         // s
      int col = (idx & 7) * 8;                    // dv local
      size_t g = (tok0 + c * 64 + row) * (size_t)2048 + h * 512 + vt * 64 + col;
      v8bf vv = *(const v8bf*)(vb + g);
#pragma unroll
      for (int e = 0; e < 8; ++e) sVT[(col + e) * 72 + row] = vv[e];
    }
    __syncthreads();

    // ---- intra scores: A = qhat @ khat^T, lower-triangular mask -> sAb ----
#pragma unroll
    for (int tt = 0; tt < 2; ++tt) {
      int tau = wave * 2 + tt;
      int ti = tau >> 2, tj = tau & 3;            // (t-tile, s-tile)
      if (tj > ti) {
#pragma unroll
        for (int r = 0; r < 8; ++r)
          sAb[(ti * 16 + r + half * 8) * 72 + tj * 16 + l16] = (bf16)0.f;
      } else {
        v8f a = {};
#pragma unroll
        for (int kk = 0; kk < 8; ++kk) {          // K = dk = 256
          const bf16* pa = sQ + (ti * 16 + l16) * 264 + kk * 32 + half * 8;
          v16bf af = cat8(*(const v8bf*)pa, *(const v8bf*)(pa + 16));
          v16bf bfm;
#pragma unroll
          for (int e = 0; e < 16; ++e)            // gather khat^T column s
            bfm[e] = sKT[(kk * 32 + 8 * half + (e & 7) + 16 * (e >> 3)) * 72 + tj * 16 + l16];
          a = WMMA_BF16(af, bfm, a);
        }
#pragma unroll
        for (int r = 0; r < 8; ++r) {
          int m = ti * 16 + r + half * 8;         // t
          int n = tj * 16 + l16;                  // s
          sAb[m * 72 + n] = (bf16)((n <= m) ? a[r] : 0.f);
        }
      }
    }
    __syncthreads();

    // ---- o = A@v + qhat@S (reads old S) ----
#pragma unroll
    for (int tt = 0; tt < 2; ++tt) {
      int tau = wave * 2 + tt;
      int ti = tau >> 2, vj = tau & 3;
      v8f a = {};
#pragma unroll
      for (int kk = 0; kk < 2; ++kk) {            // intra: K = 64
        const bf16* pa = sAb + (ti * 16 + l16) * 72 + kk * 32 + half * 8;
        v16bf af = cat8(*(const v8bf*)pa, *(const v8bf*)(pa + 16));
        const bf16* pb = sVT + (vj * 16 + l16) * 72 + kk * 32 + half * 8;
        v16bf bfm = cat8(*(const v8bf*)pb, *(const v8bf*)(pb + 16));
        a = WMMA_BF16(af, bfm, a);
      }
#pragma unroll
      for (int kk = 0; kk < 8; ++kk) {            // inter: K = dk = 256
        const bf16* pa = sQ + (ti * 16 + l16) * 264 + kk * 32 + half * 8;
        v16bf af = cat8(*(const v8bf*)pa, *(const v8bf*)(pa + 16));
        const bf16* pb = sS + (vj * 16 + l16) * 264 + kk * 32 + half * 8;
        v16bf bfm = cat8(*(const v8bf*)pb, *(const v8bf*)(pb + 16));
        a = WMMA_BF16(af, bfm, a);
      }
      int dvg = h * 512 + vt * 64 + vj * 16 + l16;
#pragma unroll
      for (int r = 0; r < 8; ++r) {
        int t = ti * 16 + r + half * 8;
        ob[(tok0 + c * 64 + t) * (size_t)2048 + dvg] = a[r];
      }
    }
    __syncthreads();                              // all reads of old S done

    // ---- state update: S = exp(gt) * (S + khat^T @ v) ----
    float egt = __expf(sGc[64]);
#pragma unroll
    for (int rr = 0; rr < 2; ++rr) {
      int di = wave * 2 + rr;                     // dk tile (wave owns 32 rows)
#pragma unroll
      for (int vj = 0; vj < 4; ++vj) {
        v8f cacc;
#pragma unroll
        for (int r = 0; r < 8; ++r)
          cacc[r] = (float)sS[(vj * 16 + l16) * 264 + di * 16 + r + half * 8];
#pragma unroll
        for (int kk = 0; kk < 2; ++kk) {          // K = s = 64
          const bf16* pa = sKT + (di * 16 + l16) * 72 + kk * 32 + half * 8;
          v16bf af = cat8(*(const v8bf*)pa, *(const v8bf*)(pa + 16));
          const bf16* pb = sVT + (vj * 16 + l16) * 72 + kk * 32 + half * 8;
          v16bf bfm = cat8(*(const v8bf*)pb, *(const v8bf*)(pb + 16));
          cacc = WMMA_BF16(af, bfm, cacc);
        }
#pragma unroll
        for (int r = 0; r < 8; ++r)
          sS[(vj * 16 + l16) * 264 + di * 16 + r + half * 8] = (bf16)(cacc[r] * egt);
      }
    }
  }
}

// ---------------------------------------------------------------------------
// FusedRMSNormSwishGate: out = RMSNorm(o)*gnorm_w * g*sigmoid(g), bf16 out
// one block per (token, head), 128 threads x 4 elems
// ---------------------------------------------------------------------------
__global__ __launch_bounds__(128)
void gate_norm_kernel(const float* __restrict__ ob, const bf16* __restrict__ gb,
                      const float* __restrict__ gw, bf16* __restrict__ outb) {
  __shared__ float red[128];
  int th = blockIdx.x;
  int t = th >> 2, h = th & 3;
  int tid = threadIdx.x;
  const float* op = ob + (size_t)t * 2048 + h * 512;
  float vals[4], ss = 0.f;
#pragma unroll
  for (int r = 0; r < 4; ++r) { float v = op[tid + r * 128]; vals[r] = v; ss += v * v; }
  red[tid] = ss;
  __syncthreads();
  for (int s = 64; s > 0; s >>= 1) {
    if (tid < s) red[tid] += red[tid + s];
    __syncthreads();
  }
  float rs = rsqrtf(red[0] * (1.0f / 512.0f) + 1e-5f);
  const bf16* gp = gb + (size_t)t * 2048 + h * 512;
  bf16* o2 = outb + (size_t)t * 2048 + h * 512;
#pragma unroll
  for (int r = 0; r < 4; ++r) {
    int i = tid + r * 128;
    float g = (float)gp[i];
    float sw = g / (1.f + __expf(-g));
    o2[i] = (bf16)(vals[r] * rs * gw[i] * sw);
  }
}

// ---------------------------------------------------------------------------
extern "C" void kernel_launch(void* const* d_in, const int* in_sizes, int n_in,
                              void* d_out, int out_size, void* d_ws, size_t ws_size,
                              hipStream_t stream) {
  (void)in_sizes; (void)n_in; (void)out_size; (void)ws_size;
  const float* x   = (const float*)d_in[0];
  const float* Wq  = (const float*)d_in[1];
  const float* Wk  = (const float*)d_in[2];
  const float* Wv  = (const float*)d_in[3];
  const float* Wg  = (const float*)d_in[4];
  const float* Wgk = (const float*)d_in[5];
  const float* bgk = (const float*)d_in[6];
  const float* Wo  = (const float*)d_in[7];
  const float* gnw = (const float*)d_in[8];
  float* out = (float*)d_out;

  const size_t M = 8192;                 // B*N tokens
  char* ws = (char*)d_ws;
  size_t off = 0;
  auto take = [&](size_t bytes) -> char* {
    char* p = ws + off;
    off += (bytes + 255) & ~(size_t)255;
    return p;
  };
  bf16*  xbf = (bf16*)take(M * 1024 * 2);
  bf16*  Wqb = (bf16*)take(1024 * 1024 * 2);
  bf16*  Wkb = (bf16*)take(1024 * 1024 * 2);
  bf16*  Wvb = (bf16*)take(1024 * 2048 * 2);
  bf16*  Wgb = (bf16*)take(1024 * 2048 * 2);
  bf16*  Wob = (bf16*)take(2048 * 1024 * 2);
  bf16*  qb  = (bf16*)take(M * 1024 * 2);
  bf16*  kb  = (bf16*)take(M * 1024 * 2);
  bf16*  vb  = (bf16*)take(M * 2048 * 2);
  bf16*  gb  = (bf16*)take(M * 2048 * 2);
  float* gkb = (float*)take(4ull * 4 * 2048 * 4);
  float* ob  = (float*)take(M * 2048 * 4);
  bf16*  gtd = (bf16*)take(M * 2048 * 2);

  auto cvt = [&](const float* in, bf16* o, int n) {
    f32_to_bf16_kernel<<<(n + 255) / 256, 256, 0, stream>>>(in, o, n);
  };
  cvt(x,  xbf, 8388608);
  cvt(Wq, Wqb, 1048576);
  cvt(Wk, Wkb, 1048576);
  cvt(Wv, Wvb, 2097152);
  cvt(Wg, Wgb, 2097152);
  cvt(Wo, Wob, 2097152);

  // projections (bf16 out)
  gemm_bf16_kernel<0><<<dim3(1024 / 128, M / 128), 256, 0, stream>>>(xbf, Wqb, qb, (int)M, 1024, 1024);
  gemm_bf16_kernel<0><<<dim3(1024 / 128, M / 128), 256, 0, stream>>>(xbf, Wkb, kb, (int)M, 1024, 1024);
  gemm_bf16_kernel<0><<<dim3(2048 / 128, M / 128), 256, 0, stream>>>(xbf, Wvb, vb, (int)M, 2048, 1024);
  gemm_bf16_kernel<0><<<dim3(2048 / 128, M / 128), 256, 0, stream>>>(xbf, Wgb, gb, (int)M, 2048, 1024);

  gk_proj_kernel<<<dim3(8192), 256, 0, stream>>>(x, Wgk, bgk, gkb);

  size_t smem = (size_t)((64 * 264 + 256 * 72 + 64 * 72 + 64 * 72 + 64 * 264) * 2 + 68 * 4);
  gla_attn_kernel<<<dim3(8, 4, 4), 256, smem, stream>>>(qb, kb, vb, gkb, ob);

  gate_norm_kernel<<<dim3(32768), 128, 0, stream>>>(ob, gb, gnw, gtd);

  // final projection (f32 out straight to d_out)
  gemm_bf16_kernel<1><<<dim3(1024 / 128, M / 128), 256, 0, stream>>>(gtd, Wob, out, (int)M, 1024, 2048);
}
